// EdgeRiskGNN_87393994539538
// MI455X (gfx1250) — compile-verified, matrix operand order checked by reading, not run on hardware
//
#include <hip/hip_runtime.h>
#include <hip/hip_bf16.h>
#include <cstdint>

#define N_NODES 50000
#define N_EDGES 800000
#define H_DIM   128
#define BN_EPS  1e-5f

typedef __bf16 bf16_t;
typedef __bf16 v16bf __attribute__((ext_vector_type(16)));
typedef __bf16 v8bf  __attribute__((ext_vector_type(8)));
typedef float  v8f   __attribute__((ext_vector_type(8)));
typedef unsigned int u32x4 __attribute__((ext_vector_type(4)));
typedef int i32x8 __attribute__((ext_vector_type(8)));
typedef int i32x4 __attribute__((ext_vector_type(4)));

#if defined(__has_builtin)
#if __has_builtin(__builtin_amdgcn_tensor_load_to_lds) && \
    __has_builtin(__builtin_amdgcn_s_wait_tensorcnt)
#define HAVE_TDM 1
#endif
#endif

// ---------------------------------------------------------------------------
// WMMA helpers (CDNA5 wave32, V_WMMA_F32_16X16X32_BF16)
// A frag (16x32): lane l holds row (l&15); kh=(l>>4):
//   elems 0..7  at K = kblk*32 + kh*8 + {0..7}; elems 8..15 at +16.
// B frag symmetric with lane = output column -> N-major bf16 weights load
// with the same two 16B reads. C/D: lane holds col (l&15), VGPR v holds
// row v + 8*(l>>4).
// ---------------------------------------------------------------------------
__device__ __forceinline__ v8f wmma_bf16(v16bf a, v16bf b, v8f c) {
  return __builtin_amdgcn_wmma_f32_16x16x32_bf16(false, a, false, b, (short)0, c,
                                                 false, false);
}

__device__ __forceinline__ v16bf frag_ld(const bf16_t* p) {
  v8bf lo = *reinterpret_cast<const v8bf*>(p);
  v8bf hi = *reinterpret_cast<const v8bf*>(p + 16);
  return __builtin_shufflevector(lo, hi, 0, 1, 2, 3, 4, 5, 6, 7, 8, 9, 10, 11,
                                 12, 13, 14, 15);
}

__device__ __forceinline__ v16bf frag_f32(const float* p, float s) {
  v16bf f;
#pragma unroll
  for (int i = 0; i < 8; ++i) f[i] = (bf16_t)(p[i] * s);
#pragma unroll
  for (int i = 0; i < 8; ++i) f[8 + i] = (bf16_t)(p[16 + i] * s);
  return f;
}

__device__ __forceinline__ float relu_f(float x) { return x > 0.f ? x : 0.f; }

// ---------------------------------------------------------------------------
// TDM staging: issue a 1-row qword-tile descriptor copy global->LDS.
// Call from one wave; drain with tdm_wait_all() before cross-wave use.
// ---------------------------------------------------------------------------
__device__ __forceinline__ void tdm_issue(void* lds_dst, const void* gsrc,
                                          unsigned nbytes) {
#if defined(HAVE_TDM)
  unsigned ldsoff = (unsigned)(uintptr_t)lds_dst;  // low 32b of LDS flat addr
  unsigned long long ga = (unsigned long long)(uintptr_t)gsrc;
  unsigned qw = nbytes >> 3;
  u32x4 g0;
  g0[0] = 1u;                                   // count=1 (valid), user mode
  g0[1] = ldsoff;                               // lds_addr  (bits 63:32)
  g0[2] = (unsigned)ga;                         // global_addr lo (95:64)
  g0[3] = (unsigned)(ga >> 32) | (2u << 30);    // global_addr hi | type=2
  i32x8 g1;
  g1[0] = (int)(3u << 16);                      // data_size=3 (8B), mask=0
  g1[1] = (int)((qw & 0xFFFFu) << 16);          // tensor_dim0[15:0]
  g1[2] = (int)((qw >> 16) | (1u << 16));       // tensor_dim0[31:16]|dim1=1
  g1[3] = (int)((qw & 0xFFFFu) << 16);          // tile_dim0 = qw
  g1[4] = 1;                                    // tile_dim1=1, tile_dim2=0
  g1[5] = (int)qw;                              // dim0_stride lo
  g1[6] = (int)((qw & 0xFFFFu) << 16);          // dim1_stride[15:0]
  g1[7] = (int)(qw >> 16);                      // dim1_stride[47:16]
  i32x4 z4 = {0, 0, 0, 0};
#if __clang_major__ >= 23
  i32x8 z8 = {0, 0, 0, 0, 0, 0, 0, 0};
  __builtin_amdgcn_tensor_load_to_lds(g0, g1, z4, z4, z8, 0);
#else
  __builtin_amdgcn_tensor_load_to_lds(g0, g1, z4, z4, 0);
#endif
#else
  unsigned l = threadIdx.x & 31;
  for (unsigned i = l; i * 4 < nbytes; i += 32)
    ((unsigned*)lds_dst)[i] = ((const unsigned*)gsrc)[i];
#endif
}

__device__ __forceinline__ void tdm_wait_all() {
#if defined(HAVE_TDM)
  __builtin_amdgcn_s_wait_tensorcnt(0);
#endif
}

// ---------------------------------------------------------------------------
// Weight prep: dst[n*stride + coloff + k] = bf16(src[k*Nd + n])
// ---------------------------------------------------------------------------
__global__ void twcvt_kernel(const float* __restrict__ src, bf16_t* __restrict__ dst,
                             int K, int Nd, int stride, int coloff) {
  int i = blockIdx.x * blockDim.x + threadIdx.x;
  if (i >= K * Nd) return;
  int n = i / K, k = i - n * K;
  dst[n * stride + coloff + k] = (bf16_t)src[k * Nd + n];
}

__global__ void zero_kernel(float* __restrict__ p, int n) {
  int i = blockIdx.x * blockDim.x + threadIdx.x;
  if (i < n) p[i] = 0.f;
}

__global__ void deg_kernel(const int* __restrict__ ei, float* __restrict__ deg) {
  int e = blockIdx.x * blockDim.x + threadIdx.x;
  if (e >= N_EDGES) return;
  atomicAdd(&deg[ei[N_EDGES + e]], 1.0f);
}

__global__ void scatter_kernel(const int* __restrict__ ei, const float* __restrict__ h,
                               float* __restrict__ agg) {
  int t = blockIdx.x * blockDim.x + threadIdx.x;
  if (t >= N_EDGES * 32) return;
  int e = t >> 5;
  int c = (t & 31) * 4;
  int s = ei[e];
  int d = ei[N_EDGES + e];
  const float* hs = h + (size_t)s * H_DIM + c;
  float* ag = agg + (size_t)d * H_DIM + c;
  atomicAdd(ag + 0, hs[0]);
  atomicAdd(ag + 1, hs[1]);
  atomicAdd(ag + 2, hs[2]);
  atomicAdd(ag + 3, hs[3]);
}

// ---------------------------------------------------------------------------
// Node encoder: h = relu(x @ W0 + b0); one wave per 16 rows.
// ---------------------------------------------------------------------------
__global__ void node_enc_kernel(const float* __restrict__ x,
                                const bf16_t* __restrict__ W0t,
                                const float* __restrict__ b0,
                                float* __restrict__ h, bf16_t* __restrict__ hbf) {
  int gw = (blockIdx.x * blockDim.x + threadIdx.x) >> 5;
  int m0 = gw * 16;
  if (m0 >= N_NODES) return;  // wave-uniform
  int l = threadIdx.x & 31;
  int col = l & 15;
  int kh = l >> 4;

  const float* xrow = x + (size_t)(m0 + col) * 64 + kh * 8;
  v16bf af[2];
  af[0] = frag_f32(xrow, 1.f);
  af[1] = frag_f32(xrow + 32, 1.f);

  v8f c[8] = {};
#pragma unroll
  for (int kt = 0; kt < 2; ++kt)
#pragma unroll
    for (int nt = 0; nt < 8; ++nt)
      c[nt] = wmma_bf16(af[kt],
                        frag_ld(W0t + (size_t)(nt * 16 + col) * 64 + kh * 8 + kt * 32),
                        c[nt]);

#pragma unroll
  for (int nt = 0; nt < 8; ++nt) {
    int n0 = nt * 16;
    float bias = b0[n0 + col];
#pragma unroll
    for (int v = 0; v < 8; ++v) {
      float t = relu_f(c[nt][v] + bias);
      size_t idx = (size_t)(m0 + kh * 8 + v) * H_DIM + n0 + col;
      h[idx] = t;
      hbf[idx] = (bf16_t)t;
    }
  }
}

// ---------------------------------------------------------------------------
// Layer GEMM: hn = [agg/deg | h] @ [Wl;Wr] + bl ; fused BN partial sums.
// Two halves of 4 n-tiles; kt-outer / nt-inner with explicit B double-buffer.
// ---------------------------------------------------------------------------
__global__ void layer_gemm_kernel(const float* __restrict__ agg,
                                  const float* __restrict__ deg,
                                  const bf16_t* __restrict__ hbf,
                                  const bf16_t* __restrict__ Wcat,
                                  const float* __restrict__ bl,
                                  float* __restrict__ hn,
                                  float* __restrict__ bnsum,
                                  float* __restrict__ bnsq) {
  int gw = (blockIdx.x * blockDim.x + threadIdx.x) >> 5;
  int m0 = gw * 16;
  if (m0 >= N_NODES) return;
  int l = threadIdx.x & 31;
  int col = l & 15;
  int kh = l >> 4;

  float dg = deg[m0 + col];
  float inv = 1.0f / (dg < 1.f ? 1.f : dg);
  const float* arow = agg + (size_t)(m0 + col) * H_DIM + kh * 8;
  const bf16_t* hrow = hbf + (size_t)(m0 + col) * H_DIM + kh * 8;

  v16bf af[8];
#pragma unroll
  for (int kt = 0; kt < 4; ++kt) af[kt] = frag_f32(arow + kt * 32, inv);
#pragma unroll
  for (int kt = 0; kt < 4; ++kt) af[4 + kt] = frag_ld(hrow + kt * 32);

#pragma unroll
  for (int half = 0; half < 2; ++half) {
    const bf16_t* wbase = Wcat + (size_t)(half * 64 + col) * 256 + kh * 8;
    v8f c[4] = {};
    v16bf bcur[4], bnxt[4];
#pragma unroll
    for (int nt = 0; nt < 4; ++nt) bcur[nt] = frag_ld(wbase + (size_t)nt * 16 * 256);
#pragma unroll
    for (int kt = 0; kt < 8; ++kt) {
      if (kt + 1 < 8) {
#pragma unroll
        for (int nt = 0; nt < 4; ++nt)
          bnxt[nt] = frag_ld(wbase + (size_t)nt * 16 * 256 + (kt + 1) * 32);
      }
#pragma unroll
      for (int nt = 0; nt < 4; ++nt) c[nt] = wmma_bf16(af[kt], bcur[nt], c[nt]);
#pragma unroll
      for (int nt = 0; nt < 4; ++nt) bcur[nt] = bnxt[nt];
    }
#pragma unroll
    for (int nt = 0; nt < 4; ++nt) {
      int n0 = (half * 4 + nt) * 16;
      float bias = bl[n0 + col];
      float s = 0.f, s2 = 0.f;
#pragma unroll
      for (int v = 0; v < 8; ++v) {
        float t = c[nt][v] + bias;
        hn[(size_t)(m0 + kh * 8 + v) * H_DIM + n0 + col] = t;
        s += t;
        s2 += t * t;
      }
      atomicAdd(&bnsum[n0 + col], s);
      atomicAdd(&bnsq[n0 + col], s2);
    }
  }
}

__global__ void bn_finalize_kernel(const float* __restrict__ bnsum,
                                   const float* __restrict__ bnsq,
                                   const float* __restrict__ gamma,
                                   const float* __restrict__ beta,
                                   float* __restrict__ scale,
                                   float* __restrict__ shift) {
  int c = threadIdx.x;
  if (c >= H_DIM) return;
  float mu = bnsum[c] * (1.0f / N_NODES);
  float var = bnsq[c] * (1.0f / N_NODES) - mu * mu;
  float sc = rsqrtf(var + BN_EPS) * gamma[c];
  scale[c] = sc;
  shift[c] = beta[c] - mu * sc;
}

__global__ void bn_apply_kernel(const float* __restrict__ hn,
                                const float* __restrict__ scale,
                                const float* __restrict__ shift,
                                float* __restrict__ h, bf16_t* __restrict__ hbf) {
  int i = blockIdx.x * blockDim.x + threadIdx.x;
  if (i >= N_NODES * H_DIM) return;
  int c = i & (H_DIM - 1);
  float t = relu_f(hn[i] * scale[c] + shift[c]);
  float nh = h[i] + t;
  h[i] = nh;
  hbf[i] = (bf16_t)nh;
}

// ---------------------------------------------------------------------------
// Fully fused edge pipeline, one wave per 16 edges:
//   ee1 = relu(ea@We1+b1)            -> bufA (bf16)
//   ee  = ee1@We2+b2                 -> bufB (bf16)
//   r1  = relu([h[s]|h[t]|ee]@Wp1+b) -> bufB (reuse; ee already in regs)
//   r2  = relu(r1@Wp2+b)             -> bufA as f32 (reuse)
//   out = sigmoid(r2@wp3+b)
// We1 + Wp2 panels staged into LDS once per block via TDM (two DMAs).
// ---------------------------------------------------------------------------
__global__ void __launch_bounds__(128) edge_pred_kernel(
    const int* __restrict__ ei, const float* __restrict__ ea,
    const bf16_t* __restrict__ hbf,
    const bf16_t* __restrict__ We1t, const float* __restrict__ be1,
    const bf16_t* __restrict__ We2t, const float* __restrict__ be2,
    const bf16_t* __restrict__ Wp1t, const float* __restrict__ bp1,
    const bf16_t* __restrict__ Wp2t, const float* __restrict__ bp2,
    const float* __restrict__ wp3, const float* __restrict__ bp3,
    float* __restrict__ out) {
  __shared__ __align__(16) bf16_t bufA[4][16 * H_DIM];   // ee1, later r2 (f32)
  __shared__ __align__(16) bf16_t bufB[4][16 * H_DIM];   // ee,  later r1
  __shared__ __align__(16) bf16_t w2s[64 * H_DIM];       // Wp2t panel (16 KB)
  __shared__ __align__(16) bf16_t w1s[128 * 32];         // We1t panel (8 KB)

  if (threadIdx.x < 32) {
    tdm_issue(&w2s[0], Wp2t, 64 * H_DIM * 2);
    tdm_issue(&w1s[0], We1t, 128 * 32 * 2);
    tdm_wait_all();
  }
  __syncthreads();

  int w = threadIdx.x >> 5;
  int e0 = (blockIdx.x * 4 + w) * 16;
  int l = threadIdx.x & 31;
  int col = l & 15;
  int kh = l >> 4;
  bf16_t* tA = &bufA[w][0];
  bf16_t* tB = &bufB[w][0];

  // --- edge encoder GEMM1: [16,32] x [32,128], B from LDS ---
  {
    v16bf a0 = frag_f32(ea + (size_t)(e0 + col) * 32 + kh * 8, 1.f);
    v8f c[8] = {};
#pragma unroll
    for (int nt = 0; nt < 8; ++nt)
      c[nt] = wmma_bf16(a0, frag_ld(&w1s[(size_t)(nt * 16 + col) * 32 + kh * 8]),
                        c[nt]);
#pragma unroll
    for (int nt = 0; nt < 8; ++nt) {
      int n0 = nt * 16;
      float bias = be1[n0 + col];
#pragma unroll
      for (int v = 0; v < 8; ++v)
        tA[(kh * 8 + v) * H_DIM + n0 + col] = (bf16_t)relu_f(c[nt][v] + bias);
    }
  }

  // --- edge encoder GEMM2: [16,128] x [128,128] ---
  {
    v16bf ef[4];
#pragma unroll
    for (int kt = 0; kt < 4; ++kt)
      ef[kt] = frag_ld(tA + col * H_DIM + kh * 8 + kt * 32);
#pragma unroll
    for (int half = 0; half < 2; ++half) {
      v8f c[4] = {};
#pragma unroll
      for (int kt = 0; kt < 4; ++kt)
#pragma unroll
        for (int nt = 0; nt < 4; ++nt) {
          int n0 = (half * 4 + nt) * 16;
          c[nt] = wmma_bf16(
              ef[kt],
              frag_ld(We2t + (size_t)(n0 + col) * H_DIM + kh * 8 + kt * 32), c[nt]);
        }
#pragma unroll
      for (int nt = 0; nt < 4; ++nt) {
        int n0 = (half * 4 + nt) * 16;
        float bias = be2[n0 + col];
#pragma unroll
        for (int v = 0; v < 8; ++v)
          tB[(kh * 8 + v) * H_DIM + n0 + col] = (bf16_t)(c[nt][v] + bias);
      }
    }
  }

  // --- gather A fragments for pred1 (K=384: src | tgt | ee) ---
  int sidx = ei[e0 + col];
  int tidx = ei[N_EDGES + e0 + col];
  const bf16_t* srow = hbf + (size_t)sidx * H_DIM + kh * 8;
  const bf16_t* trow = hbf + (size_t)tidx * H_DIM + kh * 8;
  v16bf af[12];
#pragma unroll
  for (int kt = 0; kt < 4; ++kt) af[kt] = frag_ld(srow + kt * 32);
#pragma unroll
  for (int kt = 0; kt < 4; ++kt) af[4 + kt] = frag_ld(trow + kt * 32);
#pragma unroll
  for (int kt = 0; kt < 4; ++kt)
    af[8 + kt] = frag_ld(tB + col * H_DIM + kh * 8 + kt * 32);

  // --- pred1: [16,384] x [384,128] -> r1 in bufB; B double-buffered ---
#pragma unroll
  for (int half = 0; half < 2; ++half) {
    const bf16_t* wbase = Wp1t + (size_t)(half * 64 + col) * 384 + kh * 8;
    v8f c[4] = {};
    v16bf bcur[4], bnxt[4];
#pragma unroll
    for (int nt = 0; nt < 4; ++nt) bcur[nt] = frag_ld(wbase + (size_t)nt * 16 * 384);
#pragma unroll
    for (int kt = 0; kt < 12; ++kt) {
      if (kt + 1 < 12) {
#pragma unroll
        for (int nt = 0; nt < 4; ++nt)
          bnxt[nt] = frag_ld(wbase + (size_t)nt * 16 * 384 + (kt + 1) * 32);
      }
#pragma unroll
      for (int nt = 0; nt < 4; ++nt) c[nt] = wmma_bf16(af[kt], bcur[nt], c[nt]);
#pragma unroll
      for (int nt = 0; nt < 4; ++nt) bcur[nt] = bnxt[nt];
    }
#pragma unroll
    for (int nt = 0; nt < 4; ++nt) {
      int n0 = (half * 4 + nt) * 16;
      float bias = bp1[n0 + col];
#pragma unroll
      for (int v = 0; v < 8; ++v)
        tB[(kh * 8 + v) * H_DIM + n0 + col] = (bf16_t)relu_f(c[nt][v] + bias);
    }
  }

  // --- pred2: [16,128] x [128,64] -> r2 (f32) in bufA; B from LDS ---
  float* t2 = (float*)tA;
  {
    v16bf rf[4];
#pragma unroll
    for (int kt = 0; kt < 4; ++kt)
      rf[kt] = frag_ld(tB + col * H_DIM + kh * 8 + kt * 32);
    v8f c[4] = {};
#pragma unroll
    for (int kt = 0; kt < 4; ++kt)
#pragma unroll
      for (int nt = 0; nt < 4; ++nt)
        c[nt] = wmma_bf16(
            rf[kt], frag_ld(&w2s[(size_t)(nt * 16 + col) * H_DIM + kh * 8 + kt * 32]),
            c[nt]);
#pragma unroll
    for (int nt = 0; nt < 4; ++nt) {
      int n0 = nt * 16;
      float bias = bp2[n0 + col];
#pragma unroll
      for (int v = 0; v < 8; ++v)
        t2[(kh * 8 + v) * 64 + n0 + col] = relu_f(c[nt][v] + bias);
    }
  }

  // --- pred3 + sigmoid: lanes 0..15, one edge each ---
  if (l < 16) {
    float acc = bp3[0];
#pragma unroll 8
    for (int k = 0; k < 64; ++k) acc += t2[l * 64 + k] * wp3[k];
    out[e0 + l] = 1.0f / (1.0f + __expf(-acc));
  }
}

// ---------------------------------------------------------------------------
// Host launcher
// ---------------------------------------------------------------------------
extern "C" void kernel_launch(void* const* d_in, const int* in_sizes, int n_in,
                              void* d_out, int out_size, void* d_ws, size_t ws_size,
                              hipStream_t stream) {
  const float* x = (const float*)d_in[0];
  const int* ei = (const int*)d_in[1];
  const float* ea = (const float*)d_in[2];
  const float* W0 = (const float*)d_in[3];
  const float* b0 = (const float*)d_in[4];
  const float *Wl[3], *bl[3], *Wr[3], *gam[3], *bet[3];
  for (int i = 0; i < 3; ++i) {
    int base = 5 + i * 5;
    Wl[i] = (const float*)d_in[base + 0];
    bl[i] = (const float*)d_in[base + 1];
    Wr[i] = (const float*)d_in[base + 2];
    gam[i] = (const float*)d_in[base + 3];
    bet[i] = (const float*)d_in[base + 4];
  }
  const float* We1 = (const float*)d_in[20];
  const float* be1 = (const float*)d_in[21];
  const float* We2 = (const float*)d_in[22];
  const float* be2 = (const float*)d_in[23];
  const float* Wp1 = (const float*)d_in[24];
  const float* bp1 = (const float*)d_in[25];
  const float* Wp2 = (const float*)d_in[26];
  const float* bp2 = (const float*)d_in[27];
  const float* wp3 = (const float*)d_in[28];
  const float* bp3 = (const float*)d_in[29];
  float* out = (float*)d_out;

  char* ws = (char*)d_ws;
  size_t off = 0;
  auto alloc = [&](size_t bytes) -> void* {
    void* p = ws + off;
    off = (off + bytes + 255) & ~(size_t)255;
    return p;
  };
  float* h = (float*)alloc((size_t)N_NODES * H_DIM * 4);
  bf16_t* hbf = (bf16_t*)alloc((size_t)N_NODES * H_DIM * 2);
  float* agg = (float*)alloc((size_t)N_NODES * H_DIM * 4);
  float* hn = (float*)alloc((size_t)N_NODES * H_DIM * 4);
  float* deg = (float*)alloc((size_t)N_NODES * 4);
  float* bnacc = (float*)alloc(256 * 4);  // [0..127]=sum, [128..255]=sumsq
  float* bnscale = (float*)alloc(128 * 4);
  float* bnshift = (float*)alloc(128 * 4);
  bf16_t* W0t = (bf16_t*)alloc(128 * 64 * 2);
  bf16_t* Wcat[3];
  for (int i = 0; i < 3; ++i) Wcat[i] = (bf16_t*)alloc(128 * 256 * 2);
  bf16_t* We1t = (bf16_t*)alloc(128 * 32 * 2);
  bf16_t* We2t = (bf16_t*)alloc(128 * 128 * 2);
  bf16_t* Wp1t = (bf16_t*)alloc(128 * 384 * 2);
  bf16_t* Wp2t = (bf16_t*)alloc(64 * 128 * 2);

  auto T = [&](const float* src, bf16_t* dst, int K, int Nd, int stride, int coloff) {
    int tot = K * Nd;
    twcvt_kernel<<<(tot + 255) / 256, 256, 0, stream>>>(src, dst, K, Nd, stride, coloff);
  };
  T(W0, W0t, 64, 128, 64, 0);
  for (int i = 0; i < 3; ++i) {
    T(Wl[i], Wcat[i], 128, 128, 256, 0);
    T(Wr[i], Wcat[i], 128, 128, 256, 128);
  }
  T(We1, We1t, 32, 128, 32, 0);
  T(We2, We2t, 128, 128, 128, 0);
  T(Wp1, Wp1t, 384, 128, 384, 0);
  T(Wp2, Wp2t, 128, 64, 128, 0);

  // degree
  zero_kernel<<<(N_NODES + 255) / 256, 256, 0, stream>>>(deg, N_NODES);
  deg_kernel<<<(N_EDGES + 255) / 256, 256, 0, stream>>>(ei, deg);

  // node encoder: 8 waves per 256-thread block
  {
    int waves = N_NODES / 16;  // 3125
    int blocks = (waves + 7) / 8;
    node_enc_kernel<<<blocks, 256, 0, stream>>>(x, W0t, b0, h, hbf);
  }

  // 3 SAGE layers
  for (int i = 0; i < 3; ++i) {
    zero_kernel<<<(N_NODES * H_DIM + 255) / 256, 256, 0, stream>>>(agg, N_NODES * H_DIM);
    zero_kernel<<<1, 256, 0, stream>>>(bnacc, 256);
    scatter_kernel<<<(N_EDGES * 32) / 256, 256, 0, stream>>>(ei, h, agg);
    {
      int waves = N_NODES / 16;
      int blocks = (waves + 7) / 8;
      layer_gemm_kernel<<<blocks, 256, 0, stream>>>(agg, deg, hbf, Wcat[i], bl[i], hn,
                                                    bnacc, bnacc + 128);
    }
    bn_finalize_kernel<<<1, 128, 0, stream>>>(bnacc, bnacc + 128, gam[i], bet[i],
                                              bnscale, bnshift);
    bn_apply_kernel<<<(N_NODES * H_DIM + 255) / 256, 256, 0, stream>>>(hn, bnscale,
                                                                       bnshift, h, hbf);
  }

  // fully fused edge encoder + predictor: 4 waves / 128-thread block
  edge_pred_kernel<<<N_EDGES / 64, 128, 0, stream>>>(ei, ea, hbf, We1t, be1, We2t, be2,
                                                     Wp1t, bp1, Wp2t, bp2, wp3, bp3,
                                                     out);
}
